// Head_65755949302213
// MI455X (gfx1250) — matrix operand order repaired and checked by reading, not compiled, not run on hardware
//
#include <hip/hip_runtime.h>
#include <math.h>

// ---------------------------------------------------------------------------
// Causal multi-head attention block for MI455X (gfx1250, wave32, WMMA).
// qkv = x@Wa + ba ; flash-attention (bf16 WMMA, f32 online softmax) ; out = cv@Wp + bp
// All matmuls via v_wmma_f32_16x16x32_bf16. Working set (~100MB) is L2-resident
// (192MB L2), so GEMMs stream fragments straight from global; LDS used only for
// the P-matrix C-layout -> A-layout transpose inside flash attention.
// GEMM wave tile 64x64 (16 WMMA per 16 b128 loads per k-step).
// ---------------------------------------------------------------------------

typedef __attribute__((ext_vector_type(16))) __bf16 v16bf;
typedef __attribute__((ext_vector_type(8)))  __bf16 v8bf;
typedef __attribute__((ext_vector_type(8)))  float  v8f;

#define C_EMBD 1024
#define NH     16
#define HD     64
#define BB     4
#define TT     2048
#define MTOK   (BB*TT)          // 8192 tokens
#define QKN    (3*C_EMBD)       // 3072

// log2(e) / sqrt(HD): folded into Q so softmax can use exp2
#define QSCALE 0.18033688011112042f

// Build a 16-element bf16 fragment from two 16-byte chunks.
__device__ __forceinline__ v16bf frag16(const __bf16* p0, const __bf16* p1) {
  v8bf lo = *(const v8bf*)p0;
  v8bf hi = *(const v8bf*)p1;
  return __builtin_shufflevector(lo, hi, 0,1,2,3,4,5,6,7,8,9,10,11,12,13,14,15);
}

__device__ __forceinline__ v8f wmma_bf16(v16bf a, v16bf b, v8f c) {
  // (neg_a, A, neg_b, B, c_mod, C, reuse_a, reuse_b)
  return __builtin_amdgcn_wmma_f32_16x16x32_bf16(false, a, false, b, (short)0, c,
                                                 false, false);
}

// ---------------------------------------------------------------------------
// Kernel 1: f32 -> bf16 convert (for x)
// ---------------------------------------------------------------------------
__global__ void k_f32_to_bf16(const float* __restrict__ in,
                              __bf16* __restrict__ out, int n) {
  int i = blockIdx.x * 256 + threadIdx.x;
  if (i < n) out[i] = (__bf16)in[i];
}

// ---------------------------------------------------------------------------
// Kernel 2: transpose f32 [Kd,Nd] -> bf16 [Nd,Kd]  (weights to N-major so the
// WMMA B-operand fragment is a contiguous 32B run along the K dimension)
// ---------------------------------------------------------------------------
__global__ void k_transpose_bf16(const float* __restrict__ in,
                                 __bf16* __restrict__ out, int Kd, int Nd) {
  __shared__ float tile[32][33];
  int kt = blockIdx.x * 32, nt = blockIdx.y * 32;
  int tx = threadIdx.x & 31, ty = threadIdx.x >> 5;   // 32x8
#pragma unroll
  for (int j = ty; j < 32; j += 8)
    tile[j][tx] = in[(size_t)(kt + j) * Nd + nt + tx];
  __syncthreads();
#pragma unroll
  for (int j = ty; j < 32; j += 8)
    out[(size_t)(nt + j) * Kd + kt + tx] = (__bf16)tile[tx][j];
}

// ---------------------------------------------------------------------------
// Kernel 3: QKV GEMM.  A: xb [MTOK, C] bf16 row-major.  Bt: Wa^T [3C, C] bf16.
// Block tile 128x256 (8 waves as 2Mx4N, wave tile 64x64), f32 accum; epilogue
// adds bias and scatters into Q[b,h,t,d] (pre-scaled), K[b,h,t,d], V^T[b,h,d,t].
// ---------------------------------------------------------------------------
__global__ __launch_bounds__(256)
void k_qkv_gemm(const __bf16* __restrict__ A, const __bf16* __restrict__ Bt,
                const float* __restrict__ bias,
                __bf16* __restrict__ Q, __bf16* __restrict__ K,
                __bf16* __restrict__ Vt) {
  const int lane  = threadIdx.x & 31;
  const int wave  = threadIdx.x >> 5;
  const int half  = lane >> 4;
  const int l16   = lane & 15;
  const int base0 = half ? 8 : 0;     // unused; A-frag uses kb2 too
  const int kb2   = half ? 16 : 0;    // B-frag K sub-block

  const int bm = blockIdx.x * 128 + (wave & 1) * 64;   // token rows
  const int bn = blockIdx.y * 256 + (wave >> 1) * 64;  // qkv cols

  v8f acc[4][4];
#pragma unroll
  for (int mt = 0; mt < 4; ++mt)
#pragma unroll
    for (int nt = 0; nt < 4; ++nt) acc[mt][nt] = (v8f)0.0f;

  const int Kd = C_EMBD;
  for (int k = 0; k < Kd; k += 32) {
    v16bf af[4], bf[4];
#pragma unroll
    for (int mt = 0; mt < 4; ++mt) {
      const __bf16* pa = A + (size_t)(bm + 16 * mt + l16) * Kd + k;
      af[mt] = frag16(pa + kb2, pa + kb2 + 8);
    }
#pragma unroll
    for (int nt = 0; nt < 4; ++nt) {
      const __bf16* pb = Bt + (size_t)(bn + 16 * nt + l16) * Kd + k + kb2;
      bf[nt] = frag16(pb, pb + 8);
    }
#pragma unroll
    for (int mt = 0; mt < 4; ++mt)
#pragma unroll
      for (int nt = 0; nt < 4; ++nt)
        acc[mt][nt] = wmma_bf16(af[mt], bf[nt], acc[mt][nt]);
  }

  // Epilogue: bias + scatter to Q/K/V^T layouts
#pragma unroll
  for (int mt = 0; mt < 4; ++mt)
#pragma unroll
    for (int nt = 0; nt < 4; ++nt) {
      const int n = bn + 16 * nt + l16;          // [0,3C)
      const float bn_v = bias[n];
      const int which = n >> 10;                 // 0:q 1:k 2:v
      const int c = n & (C_EMBD - 1);
      const int h = c >> 6, d = c & (HD - 1);
#pragma unroll
      for (int r = 0; r < 8; ++r) {
        const int m = bm + 16 * mt + r + 8 * half;   // token
        const int b = m >> 11, t = m & (TT - 1);
        const size_t bh = (size_t)(b * NH + h);
        const float v = acc[mt][nt][r] + bn_v;
        if (which == 0)
          Q[(bh * TT + t) * HD + d] = (__bf16)(v * QSCALE);
        else if (which == 1)
          K[(bh * TT + t) * HD + d] = (__bf16)v;
        else
          Vt[(bh * HD + d) * TT + t] = (__bf16)v;
      }
    }
}

// ---------------------------------------------------------------------------
// Kernel 4: causal flash attention.  One wave owns 32 queries of one (b,h).
// Per 32-key chunk: S = Q K^T (8 WMMA), mask (diagonal chunk only), online
// softmax (intra-half shfl_xor reductions; C-layout halves carry rows),
// P -> LDS -> A-fragments, O += P V (8 WMMA, B-frags contiguous from V^T).
// ---------------------------------------------------------------------------
__global__ __launch_bounds__(256)
void k_flash_attn(const __bf16* __restrict__ Qg, const __bf16* __restrict__ Kg,
                  const __bf16* __restrict__ Vtg, __bf16* __restrict__ CV) {
  __shared__ __bf16 plds[8][32][32];   // per-wave P tile [query][key]

  const int lane  = threadIdx.x & 31;
  const int wave  = threadIdx.x >> 5;
  const int half  = lane >> 4;
  const int l16   = lane & 15;
  const int base0 = half ? 8 : 0;
  const int kb2   = half ? 16 : 0;

  const int bh = blockIdx.y;                    // b*NH + h
  const int q0 = blockIdx.x * 256 + wave * 32;  // first query of this wave

  const __bf16* q  = Qg  + (size_t)bh * TT * HD;
  const __bf16* kk = Kg  + (size_t)bh * TT * HD;
  const __bf16* vt = Vtg + (size_t)bh * HD * TT;

  // Preload Q fragments: 2 M-tiles x 2 K-steps (hd=64)
  v16bf aq[2][2];
#pragma unroll
  for (int mt = 0; mt < 2; ++mt)
#pragma unroll
    for (int ks = 0; ks < 2; ++ks) {
      const __bf16* p = q + (size_t)(q0 + 16 * mt + l16) * HD + 32 * ks + kb2;
      aq[mt][ks] = frag16(p, p + 8);
    }

  v8f o[2][4];
  float mrow[2][8], lrow[2][8];
#pragma unroll
  for (int mt = 0; mt < 2; ++mt) {
#pragma unroll
    for (int nt = 0; nt < 4; ++nt) o[mt][nt] = (v8f)0.0f;
#pragma unroll
    for (int r = 0; r < 8; ++r) { mrow[mt][r] = -INFINITY; lrow[mt][r] = 0.0f; }
  }

  for (int kb = 0; kb <= q0; kb += 32) {        // wave-uniform trip count
    // ---- S = Q K^T for this 32-key chunk ----
    v8f s[2][2];
#pragma unroll
    for (int mt = 0; mt < 2; ++mt)
#pragma unroll
      for (int nt = 0; nt < 2; ++nt) s[mt][nt] = (v8f)0.0f;

    v16bf bk[2][2];
#pragma unroll
    for (int nt = 0; nt < 2; ++nt)
#pragma unroll
      for (int ks = 0; ks < 2; ++ks) {
        const __bf16* p = kk + (size_t)(kb + 16 * nt + l16) * HD + 32 * ks + kb2;
        bk[nt][ks] = frag16(p, p + 8);
      }
#pragma unroll
    for (int ks = 0; ks < 2; ++ks)
#pragma unroll
      for (int mt = 0; mt < 2; ++mt)
#pragma unroll
        for (int nt = 0; nt < 2; ++nt)
          s[mt][nt] = wmma_bf16(aq[mt][ks], bk[nt][ks], s[mt][nt]);

    // ---- causal mask: only the diagonal chunk needs it (kb == q0) ----
    if (kb == q0) {
#pragma unroll
      for (int mt = 0; mt < 2; ++mt)
#pragma unroll
        for (int nt = 0; nt < 2; ++nt)
#pragma unroll
          for (int r = 0; r < 8; ++r) {
            const int qrel = 16 * mt + r + 8 * half;
            const int krel = 16 * nt + l16;
            if (krel > qrel) s[mt][nt][r] = -INFINITY;
          }
    }

    // ---- online softmax (rows live on (VGPR, lane-half); keys on lanes) ----
#pragma unroll
    for (int mt = 0; mt < 2; ++mt) {
#pragma unroll
      for (int r = 0; r < 8; ++r) {
        float cm = fmaxf(s[mt][0][r], s[mt][1][r]);
#pragma unroll
        for (int d = 1; d < 16; d <<= 1)
          cm = fmaxf(cm, __shfl_xor(cm, d, 32));    // stays within half
        const float mn = fmaxf(mrow[mt][r], cm);
        const float alpha = exp2f(mrow[mt][r] - mn);
        mrow[mt][r] = mn;
        const float p0 = exp2f(s[mt][0][r] - mn);
        const float p1 = exp2f(s[mt][1][r] - mn);
        s[mt][0][r] = p0;
        s[mt][1][r] = p1;
        float rs = p0 + p1;
#pragma unroll
        for (int d = 1; d < 16; d <<= 1) rs += __shfl_xor(rs, d, 32);
        lrow[mt][r] = lrow[mt][r] * alpha + rs;
#pragma unroll
        for (int nt = 0; nt < 4; ++nt) o[mt][nt][r] *= alpha;
      }
      // P (C-layout) -> LDS [query][key] for the A-operand transpose
#pragma unroll
      for (int nt = 0; nt < 2; ++nt)
#pragma unroll
        for (int r = 0; r < 8; ++r)
          plds[wave][16 * mt + r + 8 * half][16 * nt + l16] =
              (__bf16)s[mt][nt][r];
    }

    // Same-wave LDS RAW: order + wait on DScnt (no workgroup barrier needed)
    __builtin_amdgcn_wave_barrier();
    asm volatile("s_wait_dscnt 0x0" ::: "memory");
    __builtin_amdgcn_wave_barrier();

    // ---- O += P V  (A = P from LDS, B = rows of V^T: contiguous keys) ----
    v16bf ap[2];
#pragma unroll
    for (int mt = 0; mt < 2; ++mt) {
      const __bf16* pp = &plds[wave][16 * mt + l16][kb2];
      ap[mt] = frag16(pp, pp + 8);
    }
#pragma unroll
    for (int nt = 0; nt < 4; ++nt) {
      const __bf16* pv = vt + (size_t)(16 * nt + l16) * TT + kb + kb2;
      const v16bf bv = frag16(pv, pv + 8);
#pragma unroll
      for (int mt = 0; mt < 2; ++mt)
        o[mt][nt] = wmma_bf16(ap[mt], bv, o[mt][nt]);
    }
  }

  // ---- finalize: O /= l, store into cv [B,T,C] (C = h*HD + d) ----
  const int b = bh >> 4, h = bh & 15;
#pragma unroll
  for (int mt = 0; mt < 2; ++mt)
#pragma unroll
    for (int r = 0; r < 8; ++r) {
      const float inv = 1.0f / lrow[mt][r];
      const int t = q0 + 16 * mt + r + 8 * half;
      const size_t row = (size_t)b * TT + t;
#pragma unroll
      for (int nt = 0; nt < 4; ++nt)
        CV[row * C_EMBD + h * HD + 16 * nt + l16] =
            (__bf16)(o[mt][nt][r] * inv);
    }
}

// ---------------------------------------------------------------------------
// Kernel 5: output projection GEMM + bias -> f32  (wave tile 64x64)
// ---------------------------------------------------------------------------
__global__ __launch_bounds__(256)
void k_proj_gemm(const __bf16* __restrict__ A, const __bf16* __restrict__ Bt,
                 const float* __restrict__ bias, float* __restrict__ out) {
  const int lane  = threadIdx.x & 31;
  const int wave  = threadIdx.x >> 5;
  const int half  = lane >> 4;
  const int l16   = lane & 15;
  const int base0 = half ? 8 : 0;
  const int kb2   = half ? 16 : 0;

  const int bm = blockIdx.x * 128 + (wave & 1) * 64;
  const int bn = blockIdx.y * 256 + (wave >> 1) * 64;

  v8f acc[4][4];
#pragma unroll
  for (int mt = 0; mt < 4; ++mt)
#pragma unroll
    for (int nt = 0; nt < 4; ++nt) acc[mt][nt] = (v8f)0.0f;

  const int Kd = C_EMBD;
  for (int k = 0; k < Kd; k += 32) {
    v16bf af[4], bf[4];
#pragma unroll
    for (int mt = 0; mt < 4; ++mt) {
      const __bf16* pa = A + (size_t)(bm + 16 * mt + l16) * Kd + k;
      af[mt] = frag16(pa + kb2, pa + kb2 + 8);
    }
#pragma unroll
    for (int nt = 0; nt < 4; ++nt) {
      const __bf16* pb = Bt + (size_t)(bn + 16 * nt + l16) * Kd + k + kb2;
      bf[nt] = frag16(pb, pb + 8);
    }
#pragma unroll
    for (int mt = 0; mt < 4; ++mt)
#pragma unroll
      for (int nt = 0; nt < 4; ++nt)
        acc[mt][nt] = wmma_bf16(af[mt], bf[nt], acc[mt][nt]);
  }

#pragma unroll
  for (int mt = 0; mt < 4; ++mt)
#pragma unroll
    for (int nt = 0; nt < 4; ++nt) {
      const int n = bn + 16 * nt + l16;
      const float bv = bias[n];
#pragma unroll
      for (int r = 0; r < 8; ++r) {
        const int m = bm + 16 * mt + r + 8 * half;
        out[(size_t)m * C_EMBD + n] = acc[mt][nt][r] + bv;
      }
    }
}

// ---------------------------------------------------------------------------
// Launch. Workspace layout (bf16 staging, ~88MB total):
//   xb[8192*1024] | WaT[3072*1024] | WpT[1024*1024] | Q | K | V^T | cv
// ---------------------------------------------------------------------------
extern "C" void kernel_launch(void* const* d_in, const int* in_sizes, int n_in,
                              void* d_out, int out_size, void* d_ws,
                              size_t ws_size, hipStream_t stream) {
  (void)in_sizes; (void)n_in; (void)out_size; (void)ws_size;
  const float* x      = (const float*)d_in[0];
  const float* w_attn = (const float*)d_in[1];
  const float* b_attn = (const float*)d_in[2];
  const float* w_proj = (const float*)d_in[3];
  const float* b_proj = (const float*)d_in[4];
  float* out = (float*)d_out;

  char* ws = (char*)d_ws;
  __bf16* xb  = (__bf16*)ws; ws += (size_t)MTOK * C_EMBD * 2;
  __bf16* WaT = (__bf16*)ws; ws += (size_t)QKN * C_EMBD * 2;
  __bf16* WpT = (__bf16*)ws; ws += (size_t)C_EMBD * C_EMBD * 2;
  __bf16* Qs  = (__bf16*)ws; ws += (size_t)MTOK * C_EMBD * 2;
  __bf16* Ks  = (__bf16*)ws; ws += (size_t)MTOK * C_EMBD * 2;
  __bf16* Vts = (__bf16*)ws; ws += (size_t)MTOK * C_EMBD * 2;
  __bf16* CVb = (__bf16*)ws; ws += (size_t)MTOK * C_EMBD * 2;

  // 1) x -> bf16
  {
    int n = MTOK * C_EMBD;
    k_f32_to_bf16<<<(n + 255) / 256, 256, 0, stream>>>(x, xb, n);
  }
  // 2) weight transposes (f32 [K,N] -> bf16 [N,K])
  k_transpose_bf16<<<dim3(C_EMBD / 32, QKN / 32), 256, 0, stream>>>(
      w_attn, WaT, C_EMBD, QKN);
  k_transpose_bf16<<<dim3(C_EMBD / 32, C_EMBD / 32), 256, 0, stream>>>(
      w_proj, WpT, C_EMBD, C_EMBD);
  // 3) QKV GEMM -> Q (scaled), K, V^T   (block tile 128x256)
  k_qkv_gemm<<<dim3(MTOK / 128, QKN / 256), 256, 0, stream>>>(
      xb, WaT, b_attn, Qs, Ks, Vts);
  // 4) causal flash attention -> cv bf16 [B,T,C]
  k_flash_attn<<<dim3(TT / 256, BB * NH), 256, 0, stream>>>(Qs, Ks, Vts, CVb);
  // 5) projection + bias -> f32 out   (block tile 128x256)
  k_proj_gemm<<<dim3(MTOK / 128, C_EMBD / 256), 256, 0, stream>>>(
      CVb, WpT, b_proj, out);
}
